// MaskedCrossAttention_41781441856182
// MI455X (gfx1250) — matrix operand order repaired and checked by reading, not compiled, not run on hardware
//
#include <hip/hip_runtime.h>
#include <hip/hip_bf16.h>

// ---------------------------------------------------------------------------
// Problem constants (from the reference)
// ---------------------------------------------------------------------------
#define B_ 4
#define N_ 4096
#define T_ 16
#define M_ 64
#define DIM_ 1024
#define HEADS_ 8
#define DH_ 64
#define INNER_ 512   // HEADS*DH
#define KV_ 1024     // T*M

typedef _Float16 f16;
typedef __attribute__((ext_vector_type(4)))  _Float16 v4h;
typedef __attribute__((ext_vector_type(8)))  _Float16 v8h;
typedef __attribute__((ext_vector_type(16))) _Float16 v16h;
typedef __attribute__((ext_vector_type(8)))  float    v8f;

static __device__ __forceinline__ v16h cat8(v8h a, v8h b) {
  return __builtin_shufflevector(a, b, 0,1,2,3,4,5,6,7,8,9,10,11,12,13,14,15);
}

static __device__ __forceinline__ v8f wmma_f16(v16h a, v16h b, v8f c) {
  // D = A(16x32) * B(32x16) + C, f32 accumulate (probe-verified builtin)
  return __builtin_amdgcn_wmma_f32_16x16x32_f16(false, a, false, b,
                                                (short)0, c, false, false);
}

// ---------------------------------------------------------------------------
// Weight transpose + f32 -> f16 convert:  dst[Ncols][Krows] = src[Krows][Ncols]
// ---------------------------------------------------------------------------
__global__ __launch_bounds__(256)
void transpose_convert_k(const float* __restrict__ src, f16* __restrict__ dst,
                         int Krows, int Ncols) {
  int idx = blockIdx.x * 256 + threadIdx.x;
  int total = Krows * Ncols;
  if (idx >= total) return;
  int n  = idx / Krows;
  int kk = idx - n * Krows;
  dst[idx] = (f16)src[(size_t)kk * Ncols + n];
}

// ---------------------------------------------------------------------------
// Elementwise f32 -> f16 (4 elements / thread)
// ---------------------------------------------------------------------------
__global__ __launch_bounds__(256)
void convert_f16_k(const float* __restrict__ src, f16* __restrict__ dst, int n4) {
  int idx = blockIdx.x * 256 + threadIdx.x;
  if (idx >= n4) return;
  float4 v = ((const float4*)src)[idx];
  v4h o; o.x = (f16)v.x; o.y = (f16)v.y; o.z = (f16)v.z; o.w = (f16)v.w;
  ((v4h*)dst)[idx] = o;
}

// ---------------------------------------------------------------------------
// LayerNorm over DIM=1024, one block (256 threads) per row, f16 output
// ---------------------------------------------------------------------------
__global__ __launch_bounds__(256)
void layernorm_k(const float* __restrict__ x, const float* __restrict__ g,
                 const float* __restrict__ bta, f16* __restrict__ out) {
  const int row = blockIdx.x;
  const int tid = threadIdx.x;
  const float* xr = x + (size_t)row * DIM_;
  float4 v = ((const float4*)xr)[tid];
  float s  = v.x + v.y + v.z + v.w;
  float s2 = v.x*v.x + v.y*v.y + v.z*v.z + v.w*v.w;
  #pragma unroll
  for (int off = 16; off; off >>= 1) {
    s  += __shfl_xor(s,  off, 32);
    s2 += __shfl_xor(s2, off, 32);
  }
  __shared__ float sh1[8], sh2[8];
  if ((tid & 31) == 0) { sh1[tid >> 5] = s; sh2[tid >> 5] = s2; }
  __syncthreads();
  float S = 0.f, S2 = 0.f;
  #pragma unroll
  for (int i = 0; i < 8; ++i) { S += sh1[i]; S2 += sh2[i]; }
  const float mu   = S * (1.0f / DIM_);
  const float var  = S2 * (1.0f / DIM_) - mu * mu;
  const float rstd = rsqrtf(var + 1e-5f);
  const int c = tid * 4;
  v4h o;
  o.x = (f16)((v.x - mu) * rstd * g[c+0] + bta[c+0]);
  o.y = (f16)((v.y - mu) * rstd * g[c+1] + bta[c+1]);
  o.z = (f16)((v.z - mu) * rstd * g[c+2] + bta[c+2]);
  o.w = (f16)((v.w - mu) * rstd * g[c+3] + bta[c+3]);
  *(v4h*)(out + (size_t)row * DIM_ + c) = o;
}

// ---------------------------------------------------------------------------
// Inclusive cumsum of media_locations (bool, 1 byte) -> text_time int32
// One block of 1024 threads per batch, 4 elements per thread, LDS scan.
// ---------------------------------------------------------------------------
__global__ __launch_bounds__(1024)
void cumsum_k(const unsigned char* __restrict__ loc, int* __restrict__ tt) {
  const int b = blockIdx.x, tid = threadIdx.x;
  const unsigned char* lp = loc + (size_t)b * N_ + tid * 4;
  int v0 = lp[0] ? 1 : 0, v1 = lp[1] ? 1 : 0, v2 = lp[2] ? 1 : 0, v3 = lp[3] ? 1 : 0;
  int sum = v0 + v1 + v2 + v3;
  __shared__ int sc[1024];
  sc[tid] = sum;
  __syncthreads();
  for (int off = 1; off < 1024; off <<= 1) {
    int t = (tid >= off) ? sc[tid - off] : 0;
    __syncthreads();
    sc[tid] += t;
    __syncthreads();
  }
  int run = sc[tid] - sum;  // exclusive prefix
  int* op = tt + (size_t)b * N_ + tid * 4;
  run += v0; op[0] = run;
  run += v1; op[1] = run;
  run += v2; op[2] = run;
  run += v3; op[3] = run;
}

// ---------------------------------------------------------------------------
// mean of V over kv, from transposed V layout vT[b*H+h][d][kv] (f16) -> f32
// ---------------------------------------------------------------------------
__global__ __launch_bounds__(64)
void vmean_k(const f16* __restrict__ vT, float* __restrict__ vm) {
  const int bh = blockIdx.x, d = threadIdx.x;
  const v8h* p = (const v8h*)(vT + ((size_t)bh * DH_ + d) * KV_);
  float s = 0.f;
  for (int i = 0; i < KV_ / 8; ++i) {
    v8h c = p[i];
    #pragma unroll
    for (int j = 0; j < 8; ++j) s += (float)c[j];
  }
  vm[bh * DH_ + d] = s * (1.0f / KV_);
}

// ---------------------------------------------------------------------------
// Generic f16 WMMA GEMM:  out[M x Ntot] = A[M x K] * BT[Ntot x K]^T
// Block = 256 threads (8 waves); wave computes a 32x64 tile (two M-subtiles
// sharing each B fragment in registers -> 8 WMMAs per 12 b128 loads).
// grid = (Mtot/256, Ntot/64).  mode 0: f32 row-major; 1: f16 row-major
// (*scale); 2: f16 scattered to vT[b][h][d][kv] (row = b*KV+kv, col = h*64+d).
// ---------------------------------------------------------------------------
__global__ __launch_bounds__(256)
void gemm_k(const f16* __restrict__ A, const f16* __restrict__ BT,
            void* __restrict__ outp, int K, int Ntot, int mode, float scale) {
  const int wave = threadIdx.x >> 5;
  const int lane = threadIdx.x & 31;
  const int hh  = lane >> 4;      // lane half
  const int l15 = lane & 15;
  const long Mbase = (long)blockIdx.x * 256 + wave * 32;
  const int  Nbase = blockIdx.y * 64;

  const f16* aRow0 = A + (size_t)(Mbase + l15) * K;        // rows m   = l15
  const f16* aRow1 = A + (size_t)(Mbase + 16 + l15) * K;   // rows m+16
  const f16* bBase = BT + (size_t)Nbase * K;

  v8f acc0[4] = {}, acc1[4] = {};
  for (int k0 = 0; k0 < K; k0 += 32) {
    // prefetch the A streams a few K-steps ahead (global_prefetch_b8)
    if (k0 + 128 < K) {
      __builtin_prefetch(aRow0 + k0 + 128, 0, 1);
      __builtin_prefetch(aRow1 + k0 + 128, 0, 1);
    }
    // A-matrix layout: lane half selects K sub-chunks {8h..8h+7, 16+8h..}
    v16h A0 = cat8(*(const v8h*)(aRow0 + k0 + 8 * hh),
                   *(const v8h*)(aRow0 + k0 + 16 + 8 * hh));
    v16h A1 = cat8(*(const v8h*)(aRow1 + k0 + 8 * hh),
                   *(const v8h*)(aRow1 + k0 + 16 + 8 * hh));
    #pragma unroll
    for (int nt = 0; nt < 4; ++nt) {
      // B-matrix layout: lane = column, K sub-chunks {8h..8h+7, 16+8h..}
      const f16* bp = bBase + (size_t)(nt * 16 + l15) * K + k0 + 8 * hh;
      v16h Bb = cat8(*(const v8h*)bp, *(const v8h*)(bp + 16));
      acc0[nt] = wmma_f16(A0, Bb, acc0[nt]);
      acc1[nt] = wmma_f16(A1, Bb, acc1[nt]);   // B reused from registers
    }
  }

  #pragma unroll
  for (int g = 0; g < 2; ++g) {
    v8f* acc = g ? acc1 : acc0;
    const long Mg = Mbase + g * 16;
    #pragma unroll
    for (int nt = 0; nt < 4; ++nt) {
      #pragma unroll
      for (int r = 0; r < 8; ++r) {
        const long row = Mg + r + 8 * hh;     // C layout: M = r + 8*half
        const int  col = Nbase + nt * 16 + l15;
        const float v = acc[nt][r];
        if (mode == 0) {
          ((float*)outp)[(size_t)row * Ntot + col] = v;
        } else if (mode == 1) {
          ((f16*)outp)[(size_t)row * Ntot + col] = (f16)(v * scale);
        } else {
          const int bb = (int)(row >> 10);        // row / KV
          const int kv = (int)(row & (KV_ - 1));
          const int hd = col >> 6, d = col & 63;
          ((f16*)outp)[(((size_t)bb * HEADS_ + hd) * DH_ + d) * KV_ + kv] = (f16)v;
        }
      }
    }
  }
}

// ---------------------------------------------------------------------------
// Masked cross attention.  One wave per (b, head, 16-query tile).
// text_time is monotone along n, so a 16-row tile spans few media blocks;
// flash-style online softmax over spanned blocks.  Fully-masked rows
// (tt==0 or tt>T) reproduce the reference's uniform softmax: out = mean(V).
// P is re-laid-out C->A through LDS.
// ---------------------------------------------------------------------------
__global__ __launch_bounds__(32)
void attn_k(const f16* __restrict__ q,     // [B*N][INNER], pre-scaled
            const f16* __restrict__ k,     // [B*KV][INNER]
            const f16* __restrict__ vT,    // [B*H][DH][KV]
            const float* __restrict__ vmean,// [B*H][DH]
            const int* __restrict__ tt,    // [B][N]
            f16* __restrict__ out) {       // [B*N][INNER]
  const int n0   = blockIdx.x * 16;
  const int head = blockIdx.y;
  const int b    = blockIdx.z;
  const int lane = threadIdx.x;
  const int hh   = lane >> 4;
  const int l15  = lane & 15;

  __shared__ int tts[16];
  __shared__ f16 P[16][64];

  if (lane < 16) tts[lane] = tt[(size_t)b * N_ + n0 + lane];
  __syncthreads();

  int bmin = 1 << 30, bmax = -1;
  for (int i = 0; i < 16; ++i) {
    int t = tts[i];
    if (t >= 1 && t <= T_) {
      bmin = (t - 1 < bmin) ? t - 1 : bmin;
      bmax = (t - 1 > bmax) ? t - 1 : bmax;
    }
  }

  const float* vmrow = vmean + ((size_t)b * HEADS_ + head) * DH_;

  if (bmax < 0) {  // whole tile fully masked -> uniform attention = mean(V)
    #pragma unroll
    for (int dt = 0; dt < 4; ++dt) {
      const int d = dt * 16 + l15;
      const f16 mv = (f16)vmrow[d];
      #pragma unroll
      for (int r = 0; r < 8; ++r) {
        const int Mr = r + 8 * hh;
        out[((size_t)b * N_ + n0 + Mr) * INNER_ + head * DH_ + d] = mv;
      }
    }
    return;
  }

  // Q tile as two A-matrices (K = d : 0..31, 32..63)
  const f16* qrow = q + ((size_t)b * N_ + n0 + l15) * INNER_ + head * DH_;
  v16h Aq0 = cat8(*(const v8h*)(qrow +      16 * hh), *(const v8h*)(qrow +  8 + 16 * hh));
  v16h Aq1 = cat8(*(const v8h*)(qrow + 32 + 16 * hh), *(const v8h*)(qrow + 40 + 16 * hh));

  v8f acc[4] = {};
  float mrow[8], lrow[8], scl[8];
  #pragma unroll
  for (int r = 0; r < 8; ++r) { mrow[r] = -3.0e38f; lrow[r] = 0.f; }

  for (int blk = bmin; blk <= bmax; ++blk) {
    // ---- sim = Q . K^T for this 64-wide media block (4 column tiles)
    v8f s[4];
    #pragma unroll
    for (int jt = 0; jt < 4; ++jt) {
      const f16* krow = k + ((size_t)b * KV_ + blk * 64 + jt * 16 + l15) * INNER_
                          + head * DH_;
      v16h B0 = *(const v16h*)(krow +      16 * hh);
      v16h B1 = *(const v16h*)(krow + 32 + 16 * hh);
      v8f z = {};
      z = wmma_f16(Aq0, B0, z);
      s[jt] = wmma_f16(Aq1, B1, z);
    }

    // ---- online softmax row update (rows live across 16 lanes of a half)
    #pragma unroll
    for (int r = 0; r < 8; ++r) {
      const bool rv = (tts[r + 8 * hh] == blk + 1);
      float smax = -3.0e38f;
      #pragma unroll
      for (int jt = 0; jt < 4; ++jt) smax = fmaxf(smax, s[jt][r]);
      #pragma unroll
      for (int off = 8; off; off >>= 1) smax = fmaxf(smax, __shfl_xor(smax, off, 32));
      if (rv) {
        const float mnew = fmaxf(mrow[r], smax);
        const float sc   = __expf(mrow[r] - mnew);
        float ps = 0.f;
        #pragma unroll
        for (int jt = 0; jt < 4; ++jt) {
          float p = __expf(s[jt][r] - mnew);
          s[jt][r] = p;
          ps += p;
        }
        #pragma unroll
        for (int off = 8; off; off >>= 1) ps += __shfl_xor(ps, off, 32);
        lrow[r] = lrow[r] * sc + ps;
        mrow[r] = mnew;
        scl[r]  = sc;
      } else {
        #pragma unroll
        for (int jt = 0; jt < 4; ++jt) s[jt][r] = 0.f;
        scl[r] = 1.0f;
      }
    }

    // ---- C-layout P  ->  LDS  ->  A-layout P
    __syncthreads();
    #pragma unroll
    for (int jt = 0; jt < 4; ++jt)
      #pragma unroll
      for (int r = 0; r < 8; ++r)
        P[r + 8 * hh][jt * 16 + l15] = (f16)s[jt][r];
    __syncthreads();

    const f16* pr = &P[l15][0];
    v16h Pa0 = cat8(*(const v8h*)(pr +      16 * hh), *(const v8h*)(pr +  8 + 16 * hh));
    v16h Pa1 = cat8(*(const v8h*)(pr + 32 + 16 * hh), *(const v8h*)(pr + 40 + 16 * hh));

    // ---- out = out*scale + P . V   (V transposed: contiguous along kv)
    #pragma unroll
    for (int dt = 0; dt < 4; ++dt) {
      #pragma unroll
      for (int r = 0; r < 8; ++r) acc[dt][r] *= scl[r];
      const f16* vrow = vT + (((size_t)b * HEADS_ + head) * DH_ + dt * 16 + l15) * KV_
                           + blk * 64;
      v16h Bv0 = *(const v16h*)(vrow +      16 * hh);
      v16h Bv1 = *(const v16h*)(vrow + 32 + 16 * hh);
      acc[dt] = wmma_f16(Pa0, Bv0, acc[dt]);
      acc[dt] = wmma_f16(Pa1, Bv1, acc[dt]);
    }
  }

  // ---- store: valid rows get acc/l, fully-masked rows get mean(V)
  #pragma unroll
  for (int dt = 0; dt < 4; ++dt) {
    const int d = dt * 16 + l15;
    const float mv = vmrow[d];
    #pragma unroll
    for (int r = 0; r < 8; ++r) {
      const int Mr = r + 8 * hh;
      const int t = tts[Mr];
      const float val = (t >= 1 && t <= T_) ? acc[dt][r] / lrow[r] : mv;
      out[((size_t)b * N_ + n0 + Mr) * INNER_ + head * DH_ + d] = (f16)val;
    }
  }
}

// ---------------------------------------------------------------------------
// Host launcher
// ---------------------------------------------------------------------------
extern "C" void kernel_launch(void* const* d_in, const int* in_sizes, int n_in,
                              void* d_out, int out_size, void* d_ws, size_t ws_size,
                              hipStream_t stream) {
  const float* x      = (const float*)d_in[0];
  const float* media  = (const float*)d_in[1];
  const unsigned char* mloc = (const unsigned char*)d_in[2];  // bool array
  const float* ln_s   = (const float*)d_in[3];
  const float* ln_b   = (const float*)d_in[4];
  const float* Wq     = (const float*)d_in[5];
  const float* Wk     = (const float*)d_in[6];
  const float* Wv     = (const float*)d_in[7];
  const float* Wo     = (const float*)d_in[8];

  char* ws = (char*)d_ws;
  size_t off = 0;
  auto alloc = [&](size_t bytes) -> void* {
    void* p = ws + off;
    off += (bytes + 255) & ~(size_t)255;
    return p;
  };

  f16* WqT   = (f16*)alloc((size_t)INNER_ * DIM_ * 2);   // [inner][dim]
  f16* WkT   = (f16*)alloc((size_t)INNER_ * DIM_ * 2);
  f16* WvT   = (f16*)alloc((size_t)INNER_ * DIM_ * 2);
  f16* WoT   = (f16*)alloc((size_t)DIM_ * INNER_ * 2);   // [dim][inner]
  f16* medh  = (f16*)alloc((size_t)B_ * KV_ * DIM_ * 2);
  f16* xnh   = (f16*)alloc((size_t)B_ * N_ * DIM_ * 2);
  int* tt    = (int*)alloc((size_t)B_ * N_ * 4);
  f16* qh    = (f16*)alloc((size_t)B_ * N_ * INNER_ * 2);
  f16* kh    = (f16*)alloc((size_t)B_ * KV_ * INNER_ * 2);
  f16* vTh   = (f16*)alloc((size_t)B_ * HEADS_ * DH_ * KV_ * 2);
  float* vmn = (float*)alloc((size_t)B_ * HEADS_ * DH_ * 4);
  f16* aout  = (f16*)alloc((size_t)B_ * N_ * INNER_ * 2);
  (void)ws_size; (void)in_sizes; (void)n_in; (void)out_size;

  // 1) weights -> f16 transposed
  transpose_convert_k<<<(DIM_*INNER_ + 255)/256, 256, 0, stream>>>(Wq, WqT, DIM_, INNER_);
  transpose_convert_k<<<(DIM_*INNER_ + 255)/256, 256, 0, stream>>>(Wk, WkT, DIM_, INNER_);
  transpose_convert_k<<<(DIM_*INNER_ + 255)/256, 256, 0, stream>>>(Wv, WvT, DIM_, INNER_);
  transpose_convert_k<<<(INNER_*DIM_ + 255)/256, 256, 0, stream>>>(Wo, WoT, INNER_, DIM_);

  // 2) media -> f16 ; LayerNorm(x) -> f16 ; cumsum -> text_time
  convert_f16_k<<<(B_*KV_*DIM_/4 + 255)/256, 256, 0, stream>>>(media, medh, B_*KV_*DIM_/4);
  layernorm_k<<<B_*N_, 256, 0, stream>>>(x, ln_s, ln_b, xnh);
  cumsum_k<<<B_, 1024, 0, stream>>>(mloc, tt);

  // 3) projections (f16 WMMA, f32 accumulate); block tile 256x64, wave 32x64
  //    q pre-scaled by DH^-0.5 = 0.125
  gemm_k<<<dim3(B_*N_/256,  INNER_/64), 256, 0, stream>>>(xnh,  WqT, qh,  DIM_, INNER_, 1, 0.125f);
  gemm_k<<<dim3(B_*KV_/256, INNER_/64), 256, 0, stream>>>(medh, WkT, kh,  DIM_, INNER_, 1, 1.0f);
  gemm_k<<<dim3(B_*KV_/256, INNER_/64), 256, 0, stream>>>(medh, WvT, vTh, DIM_, INNER_, 2, 1.0f);

  // 4) mean(V) for fully-masked rows
  vmean_k<<<B_*HEADS_, DH_, 0, stream>>>(vTh, vmn);

  // 5) masked attention
  attn_k<<<dim3(N_/16, HEADS_, B_), 32, 0, stream>>>(qh, kh, vTh, vmn, tt, aout);

  // 6) output projection -> f32 d_out
  gemm_k<<<dim3(B_*N_/256, DIM_/64), 256, 0, stream>>>(aout, WoT, d_out, INNER_, DIM_, 0, 1.0f);
}
